// T5Attention_62302795596673
// MI455X (gfx1250) — compile-verified, matrix operand order checked by reading, not run on hardware
//
#include <hip/hip_runtime.h>

// ---------------------------------------------------------------------------
// T5 attention on gfx1250 (MI455X): bf16 WMMA everywhere, f32 accumulate.
// Intermediates (q, k, vT) stored as bf16; V stored transposed so every WMMA
// fragment load in attention is two contiguous 16-byte loads. Relative-
// position bias folded into a per-block LDS table indexed by (s - t).
// ---------------------------------------------------------------------------

typedef __attribute__((ext_vector_type(16))) __bf16        v16bf;
typedef __attribute__((ext_vector_type(8)))  float         v8f;
typedef __attribute__((ext_vector_type(8)))  unsigned int  v8u32;

#define T5_B 2
#define T5_S 2048
#define T5_D 1024
#define T5_H 16
#define T5_DK 64
#define T5_INNER 1024

// ----- bf16 helpers (RNE) ---------------------------------------------------
__device__ __forceinline__ unsigned short f2bf_bits(float f) {
  unsigned u = __builtin_bit_cast(unsigned, f);
  u += 0x7FFFu + ((u >> 16) & 1u);
  return (unsigned short)(u >> 16);
}

// CDNA5 ISA 7.12.2, 16-bit A/B 16x32 fragment: lanes 0-15 hold K {0..7,16..23},
// lanes 16-31 hold K {8..15,24..31}; VGPR j holds K pair (k,k+1). Each half is
// 8 contiguous bf16 = 16 bytes, so a fragment is two b128 loads.
__device__ __forceinline__ v16bf fragBF16(const unsigned short* rowPtr, int lane) {
  const int base = (lane & 16) ? 8 : 0;
  uint4 lo = *(const uint4*)(rowPtr + base);        // K = base .. base+7
  uint4 hi = *(const uint4*)(rowPtr + 16 + base);   // K = 16+base .. 16+base+7
  v8u32 t;
  t[0] = lo.x; t[1] = lo.y; t[2] = lo.z; t[3] = lo.w;
  t[4] = hi.x; t[5] = hi.y; t[6] = hi.z; t[7] = hi.w;
  return __builtin_bit_cast(v16bf, t);
}

__device__ __forceinline__ v8f wmma_bf16(v16bf a, v16bf b, v8f c) {
  return __builtin_amdgcn_wmma_f32_16x16x32_bf16(
      /*neg_a=*/false, a, /*neg_b=*/false, b,
      /*c_mod=*/(short)0, c, /*reuse_a=*/false, /*reuse_b=*/false);
}

// T5 relative position bucket (bidirectional, 32 buckets, max_distance 128).
// rel = memory_pos - context_pos
__device__ __forceinline__ int relBucket(int rel) {
  int bucket = (rel > 0) ? 16 : 0;
  int r = (rel < 0) ? -rel : rel;
  if (r < 8) {
    bucket += r;
  } else {
    float v = 8.0f + __logf((float)r * 0.125f) * (8.0f / __logf(16.0f));
    v = fminf(v, 15.0f);
    bucket += (int)v;
  }
  return bucket;
}

// ---------------------------------------------------------------------------
// GEMM: C[M,N] = A[M,K] * B[K,N] * scale.  M=4096, N=K=1024 for all uses.
// Block tile 64x128, 8 waves of 32x32, K staged 64-deep in LDS as bf16.
// MODE: 0 = f32 output, 1 = bf16 output, 2 = bf16 output in transposed
//       V layout vT[b][h][d][s]  (requires S=2048, H=16, Dk=64 shape).
// ---------------------------------------------------------------------------
#define GBM 64
#define GBN 128
#define GKC 64

template <int MODE>
__global__ __launch_bounds__(256)
void t5_gemm_bf16_wmma(const float* __restrict__ A, const float* __restrict__ B,
                       void* __restrict__ Cout, int M, int N, int K, float scale) {
  __shared__ unsigned short As[GBM][GKC];  //  8 KB, row-major (M x K)
  __shared__ unsigned short Bs[GBN][GKC];  // 16 KB, N-major (transposed tile)

  const int tid  = threadIdx.x;
  const int lane = tid & 31;
  const int wave = tid >> 5;
  const int mBlk = blockIdx.y * GBM;
  const int nBlk = blockIdx.x * GBN;
  const int wm   = (wave >> 2) * 32;  // 2 waves along M
  const int wn   = (wave & 3) * 32;   // 4 waves along N

  v8f acc[2][2] = {};

  for (int k0 = 0; k0 < K; k0 += GKC) {
    {  // stage A: 64 rows x 64 k, 16 consecutive f32 per thread
      int row = tid >> 2;
      int kk  = (tid & 3) * 16;
      const float* src = A + (size_t)(mBlk + row) * K + k0 + kk;
#pragma unroll
      for (int i = 0; i < 16; ++i) As[row][kk + i] = f2bf_bits(src[i]);
      if (k0 + GKC < K) __builtin_prefetch(src + GKC, 0, 0);
    }
    {  // stage B transposed: coalesced over n, strided LDS write
      int n  = tid & 127;
      int ks = tid >> 7;
#pragma unroll
      for (int i = 0; i < 32; ++i) {
        int k = ks + 2 * i;
        Bs[n][k] = f2bf_bits(B[(size_t)(k0 + k) * N + nBlk + n]);
      }
      if (k0 + GKC < K)
        __builtin_prefetch(B + (size_t)(k0 + GKC + ks) * N + nBlk + n, 0, 0);
    }
    __syncthreads();

#pragma unroll
    for (int ks = 0; ks < GKC; ks += 32) {
      v16bf a0 = fragBF16(&As[wm + (lane & 15)][ks], lane);
      v16bf a1 = fragBF16(&As[wm + 16 + (lane & 15)][ks], lane);
      v16bf b0 = fragBF16(&Bs[wn + (lane & 15)][ks], lane);
      v16bf b1 = fragBF16(&Bs[wn + 16 + (lane & 15)][ks], lane);
      acc[0][0] = wmma_bf16(a0, b0, acc[0][0]);
      acc[0][1] = wmma_bf16(a0, b1, acc[0][1]);
      acc[1][0] = wmma_bf16(a1, b0, acc[1][0]);
      acc[1][1] = wmma_bf16(a1, b1, acc[1][1]);
    }
    __syncthreads();
  }

  // C/D layout: VGPR i -> M = i (+8 for lanes 16-31), N = lane%16
  const int mh  = (lane >> 4) * 8;
  const int nIn = lane & 15;
#pragma unroll
  for (int sm = 0; sm < 2; ++sm)
#pragma unroll
    for (int sn = 0; sn < 2; ++sn)
#pragma unroll
      for (int i = 0; i < 8; ++i) {
        int r = mBlk + wm + sm * 16 + mh + i;
        int c = nBlk + wn + sn * 16 + nIn;
        float val = acc[sm][sn][i] * scale;
        if constexpr (MODE == 0) {
          ((float*)Cout)[(size_t)r * N + c] = val;
        } else if constexpr (MODE == 1) {
          ((unsigned short*)Cout)[(size_t)r * N + c] = f2bf_bits(val);
        } else {
          // r = b*S + s ; c = h*Dk + d ; write vT[b][h][d][s]
          int bb = r >> 11, s = r & (T5_S - 1);
          int hh = c >> 6, d = c & (T5_DK - 1);
          ((unsigned short*)Cout)[(((size_t)(bb * T5_H + hh) * T5_DK + d) << 11) + s] =
              f2bf_bits(val);
        }
      }
}

// ---------------------------------------------------------------------------
// Attention: one block per (b, h, 16-query tile). Full 16x2048 score panel in
// LDS (MI455X: 320KB/WGP), 2-pass softmax, WMMA for QK^T and PV. Relative
// bias precomputed per block into biasTab[rel + 2047] (4096 entries).
// ---------------------------------------------------------------------------
#define SCORE_BYTES (16 * T5_S * 4)                     // 131072
#define P_BYTES     (16 * T5_S * 2)                     // 65536
#define BIASTAB_N   4096
#define ATTN_SHMEM  (SCORE_BYTES + P_BYTES + BIASTAB_N * 4 + 256 * 4 + 1024 * 4)

__global__ __launch_bounds__(256)
void t5_attn_wmma(const unsigned short* __restrict__ q,
                  const unsigned short* __restrict__ k,
                  const unsigned short* __restrict__ vT,
                  const int* __restrict__ mask,
                  const float* __restrict__ relBias, float* __restrict__ ctx) {
  extern __shared__ char smem[];
  float*          scores  = (float*)smem;                            // [16][2048] f32
  unsigned short* P       = (unsigned short*)(smem + SCORE_BYTES);   // [16][2048] bf16
  float*          biasTab = (float*)(smem + SCORE_BYTES + P_BYTES);  // [4096]
  float*          red     = biasTab + BIASTAB_N;                     // [16][16]
  float*          part    = red + 256;                               // [4][256]

  const int tid  = threadIdx.x;
  const int lane = tid & 31;
  const int wave = tid >> 5;

  const int tTile = blockIdx.x & 127;          // S/16 tiles
  const int h     = (blockIdx.x >> 7) & 15;
  const int b     = blockIdx.x >> 11;
  const int tBase = tTile * 16;

  // Precompute bias by relative offset: rel = j - 2047 for j in [0, 4095].
  // One transcendental pass per block instead of per score tile.
  for (int j = tid; j < BIASTAB_N; j += 256)
    biasTab[j] = relBias[relBucket(j - 2047) * T5_H + h];
  __syncthreads();

  const int m  = lane & 15;
  const int mh = (lane >> 4) * 8;

  // Q fragments for this tile's 16 rows (q already scaled by 1/sqrt(Dk))
  const unsigned short* qrow = q + ((size_t)(b * T5_S + tBase + m) * T5_H + h) * T5_DK;
  v16bf aq0 = fragBF16(qrow, lane);
  v16bf aq1 = fragBF16(qrow + 32, lane);

  // ---- scores = QK^T + bias + mask, written to LDS ----
  for (int chunk = wave; chunk < T5_S / 16; chunk += 8) {
    const int kb = chunk * 16;
    const int s  = kb + (lane & 15);  // this lane's key column
    const unsigned short* krow = k + ((size_t)(b * T5_S + s) * T5_H + h) * T5_DK;
    v16bf bk0 = fragBF16(krow, lane);
    v16bf bk1 = fragBF16(krow + 32, lane);

    v8f acc = {};
    acc = wmma_bf16(aq0, bk0, acc);
    acc = wmma_bf16(aq1, bk1, acc);

    const float maskAdd = (mask[b * T5_S + s] == 0) ? -1e30f : 0.0f;
    const int   biBase  = s - tBase - mh + 2047;  // biasTab index for i = 0
#pragma unroll
    for (int i = 0; i < 8; ++i) {
      scores[(mh + i) * T5_S + s] = acc[i] + biasTab[biBase - i] + maskAdd;
    }
  }
  __syncthreads();

  // ---- softmax: 16 threads per row, each owns 128 strided columns ----
  {
    const int r = tid & 15;
    const int g = tid >> 4;
    float mx = -3.4e38f;
#pragma unroll 4
    for (int j = 0; j < 128; ++j)
      mx = fmaxf(mx, scores[r * T5_S + g + 16 * j]);
    red[r * 16 + g] = mx;
    __syncthreads();
    float rm = red[r * 16];
#pragma unroll
    for (int jj = 1; jj < 16; ++jj) rm = fmaxf(rm, red[r * 16 + jj]);
    __syncthreads();

    float sum = 0.0f;
#pragma unroll 4
    for (int j = 0; j < 128; ++j) {
      int idx = r * T5_S + g + 16 * j;
      float e = __expf(scores[idx] - rm);
      scores[idx] = e;
      sum += e;
    }
    red[r * 16 + g] = sum;
    __syncthreads();
    float rs = 0.0f;
#pragma unroll
    for (int jj = 0; jj < 16; ++jj) rs += red[r * 16 + jj];
    float inv = 1.0f / rs;
#pragma unroll 4
    for (int j = 0; j < 128; ++j) {
      int idx = r * T5_S + g + 16 * j;
      P[idx] = f2bf_bits(scores[idx] * inv);
    }
  }
  __syncthreads();

  // ---- out = P @ V : waves split 4 N-tiles x 2 K-halves ----
  {
    const int nT = wave & 3;
    const int kh = wave >> 2;
    const int d  = nT * 16 + (lane & 15);  // this lane's output dim column
    // vT[b][h][d][s]: key dimension contiguous per (head, dim) row
    const unsigned short* vrow =
        vT + (((size_t)(b * T5_H + h) * T5_DK + d) << 11);

    v8f acc = {};
    const int kBeg = kh * (T5_S / 2);
    for (int kk = kBeg; kk < kBeg + T5_S / 2; kk += 32) {
      v16bf ap = fragBF16(&P[m * T5_S + kk], lane);
      v16bf bv = fragBF16(vrow + kk, lane);
      acc = wmma_bf16(ap, bv, acc);
    }

    if (kh == 1) {
#pragma unroll
      for (int i = 0; i < 8; ++i) part[nT * 256 + i * 32 + lane] = acc[i];
    }
    __syncthreads();
    if (kh == 0) {
#pragma unroll
      for (int i = 0; i < 8; ++i) {
        float val = acc[i] + part[nT * 256 + i * 32 + lane];
        int t = tBase + mh + i;
        ctx[(size_t)(b * T5_S + t) * T5_INNER + h * T5_DK + d] = val;
      }
    }
  }
}

// ---------------------------------------------------------------------------
extern "C" void kernel_launch(void* const* d_in, const int* in_sizes, int n_in,
                              void* d_out, int out_size, void* d_ws, size_t ws_size,
                              hipStream_t stream) {
  const float* hs      = (const float*)d_in[0];
  const int*   mask    = (const int*)d_in[1];
  const float* Wq      = (const float*)d_in[2];
  const float* Wk      = (const float*)d_in[3];
  const float* Wv      = (const float*)d_in[4];
  const float* Wo      = (const float*)d_in[5];
  const float* relBias = (const float*)d_in[6];
  float*       out     = (float*)d_out;

  const int M = T5_B * T5_S;  // 4096
  const int N = T5_INNER;     // 1024
  const int K = T5_D;         // 1024

  size_t nElem = (size_t)M * T5_INNER;
  unsigned short* dq  = (unsigned short*)d_ws;   // bf16 [B,S,H,Dk]
  unsigned short* dk  = dq + nElem;              // bf16 [B,S,H,Dk]
  unsigned short* dvT = dk + nElem;              // bf16 [B,H,Dk,S]
  float*          dctx = (float*)(dvT + nElem);  // f32  [B,S,H*Dk]

  dim3 gg(N / GBN, M / GBM);  // (8, 64)
  // Q/K/V projections: bf16 outputs, V transposed; fold 1/sqrt(Dk) into Q.
  t5_gemm_bf16_wmma<1><<<gg, 256, 0, stream>>>(hs, Wq, dq, M, N, K, 0.125f);
  t5_gemm_bf16_wmma<1><<<gg, 256, 0, stream>>>(hs, Wk, dk, M, N, K, 1.0f);
  t5_gemm_bf16_wmma<2><<<gg, 256, 0, stream>>>(hs, Wv, dvT, M, N, K, 1.0f);

  // Attention: B*H*(S/16) = 4096 blocks, ~218 KB dynamic LDS each
  hipFuncSetAttribute(reinterpret_cast<const void*>(t5_attn_wmma),
                      hipFuncAttributeMaxDynamicSharedMemorySize, ATTN_SHMEM);
  t5_attn_wmma<<<T5_B * T5_H * (T5_S / 16), 256, ATTN_SHMEM, stream>>>(
      dq, dk, dvT, mask, relBias, dctx);

  // Output projection (f32 result)
  t5_gemm_bf16_wmma<0><<<gg, 256, 0, stream>>>(dctx, Wo, out, M, N, K, 1.0f);
}
